// EnergyInstrumentModel_58961311040117
// MI455X (gfx1250) — matrix operand order repaired and checked by reading, not compiled, not run on hardware
//
#include <hip/hip_runtime.h>
#include <hip/hip_bf16.h>
#include <cstdint>
#include <cstddef>

// ---------- types ----------
typedef __bf16 bhalf;
typedef __attribute__((ext_vector_type(16))) __bf16 v16bf;
typedef __attribute__((ext_vector_type(8)))  __bf16 v8bf;
typedef __attribute__((ext_vector_type(8)))  float  v8f;

static __device__ __forceinline__ v8f wmma_bf16(v16bf a, v16bf b, v8f c) {
  return __builtin_amdgcn_wmma_f32_16x16x32_bf16(false, a, false, b, (short)0, c,
                                                 false, false);
}

static __device__ __forceinline__ v8f zero8() {
  v8f z;
#pragma unroll
  for (int i = 0; i < 8; ++i) z[i] = 0.0f;
  return z;
}

// A operand: 16x32 bf16 tile, row-major source with row stride `ld`.
// Lane L<16: row=L, K in {0..7} and {16..23}; lane>=16: row=L-16, K {8..15},{24..31}.
static __device__ __forceinline__ v16bf load_A16x32(const bhalf* base, int ld, int lane) {
  const int row = lane & 15;
  const int ko  = (lane >> 4) * 8;
  const bhalf* p = base + row * ld + ko;
  v8bf lo = *(const v8bf*)p;
  v8bf hi = *(const v8bf*)(p + 16);
  v16bf a;
#pragma unroll
  for (int i = 0; i < 8; ++i) { a[i] = lo[i]; a[i + 8] = hi[i]; }
  return a;
}

// B operand: 32x16 bf16 tile from an (N,K)-row-major matrix with row stride `ld`.
// Lane L<16: n=L, K {0..7},{16..23}; lane>=16: n=L-16, K {8..15},{24..31}.
static __device__ __forceinline__ v16bf load_B32x16(const bhalf* base, int ld, int lane) {
  const int n  = lane & 15;
  const int ko = (lane >> 4) * 8;
  const bhalf* p = base + n * ld + ko;
  v8bf lo = *(const v8bf*)p;
  v8bf hi = *(const v8bf*)(p + 16);
  v16bf b;
#pragma unroll
  for (int i = 0; i < 8; ++i) { b[i] = lo[i]; b[i + 8] = hi[i]; }
  return b;
}

// ---------- fp32 -> bf16 convert ----------
__global__ void __launch_bounds__(256) cvt_bf16_kernel(const float* __restrict__ in,
                                                       bhalf* __restrict__ out, int n) {
  int i = blockIdx.x * 256 + threadIdx.x;
  if (i < n) out[i] = (bhalf)in[i];
}

// ---------- generic GEMM: out(M x N) = A(M x K) @ W(N x K)^T ----------
// OMODE: 0 = bf16 row-major, 1 = f32 row-major, 2 = bf16 transposed store to
// vT[((batch*256 + c) << 12) + n] with batch = m>>12, n = m&4095 (nblk = 4096).
template <int NTILES, int KCH, int OMODE>
__global__ void __launch_bounds__(256) gemm_bf16_kernel(const bhalf* __restrict__ A,
                                                        const bhalf* __restrict__ W,
                                                        void* __restrict__ out) {
  const int lane = threadIdx.x & 31;
  const int wave = threadIdx.x >> 5;
  const int K    = KCH * 32;
  const int m0   = blockIdx.x * 128 + wave * 16;

  v8f acc[NTILES];
#pragma unroll
  for (int i = 0; i < NTILES; ++i) acc[i] = zero8();

#pragma unroll
  for (int kc = 0; kc < KCH; ++kc) {
    v16bf a = load_A16x32(A + (size_t)m0 * K + kc * 32, K, lane);
#pragma unroll
    for (int nt = 0; nt < NTILES; ++nt) {
      v16bf b = load_B32x16(W + (size_t)(nt * 16) * K + kc * 32, K, lane);
      acc[nt] = wmma_bf16(a, b, acc[nt]);
    }
  }

  const int mrow = (lane >> 4) * 8;  // D layout: elem j -> (m = mrow+j, n = ncol)
  const int ncol = lane & 15;
  if constexpr (OMODE == 0) {
    bhalf* o = (bhalf*)out;
#pragma unroll
    for (int nt = 0; nt < NTILES; ++nt)
#pragma unroll
      for (int j = 0; j < 8; ++j)
        o[(size_t)(m0 + mrow + j) * (NTILES * 16) + nt * 16 + ncol] = (bhalf)acc[nt][j];
  } else if constexpr (OMODE == 1) {
    float* o = (float*)out;
#pragma unroll
    for (int nt = 0; nt < NTILES; ++nt)
#pragma unroll
      for (int j = 0; j < 8; ++j)
        o[(size_t)(m0 + mrow + j) * (NTILES * 16) + nt * 16 + ncol] = acc[nt][j];
  } else {
    bhalf* o = (bhalf*)out;
#pragma unroll
    for (int nt = 0; nt < NTILES; ++nt) {
      const int c = nt * 16 + ncol;
#pragma unroll
      for (int j = 0; j < 8; ++j) {
        const int m = m0 + mrow + j;
        o[(size_t)(((m >> 12) * 256 + c) << 12) + (m & 4095)] = (bhalf)acc[nt][j];
      }
    }
  }
}

// ---------- fused masked attention-like block ----------
// out[b,s,c] = tanh( sum_{t>=s} (q_s . k_t) * v[t,c] )
// k,q: (B, 4096, 256) bf16 row-major; vT: (B, 256, 4096) bf16; hout: (B,4096,256) bf16.
__global__ void __launch_bounds__(256) attn_kernel(const bhalf* __restrict__ kin,
                                                   const bhalf* __restrict__ qin,
                                                   const bhalf* __restrict__ vT,
                                                   bhalf* __restrict__ hout) {
  constexpr int NB = 4096, C = 256;
  __shared__ __align__(16) bhalf qlds[8][16 * 256];  // 64 KB: per-wave q tile
  __shared__ __align__(16) bhalf stage[8][16 * 32];  // 8 KB: per-wave P staging

  const int lane  = threadIdx.x & 31;
  const int wave  = threadIdx.x >> 5;
  const int batch = blockIdx.y;
  const int s0    = blockIdx.x * 128 + wave * 16;

  const bhalf* kb = kin + (size_t)batch * NB * C;
  const bhalf* qb = qin + (size_t)batch * NB * C;
  const bhalf* vb = vT + (size_t)batch * C * NB;

  // stage this wave's 16x256 q tile into LDS (row-major)
  {
    const int row = lane & 15;
    const int cb  = (lane >> 4) * 128;
#pragma unroll
    for (int i = 0; i < 8; ++i) {
      const int col = cb + i * 16;
      *(v8bf*)&qlds[wave][row * 256 + col] =
          *(const v8bf*)(qb + (size_t)(s0 + row) * C + col);
    }
  }
  asm volatile("s_wait_dscnt 0" ::: "memory");

  v8f acc[16];
#pragma unroll
  for (int i = 0; i < 16; ++i) acc[i] = zero8();

  const int mrow = (lane >> 4) * 8;
  const int ncol = lane & 15;
  const int arow = lane & 15;        // A-layout addressing for LDS reads
  const int akof = (lane >> 4) * 8;

  const int t0 = s0 & ~31;
  for (int t = t0; t < NB; t += 32) {
    // ---- P[s][t'] = q_s . k_t  (two 16x16 tiles over t chunk of 32) ----
    v8f P0 = zero8(), P1 = zero8();
#pragma unroll
    for (int cc = 0; cc < 8; ++cc) {
      v16bf a;
      {
        const bhalf* sp = &qlds[wave][arow * 256 + cc * 32 + akof];
        v8bf lo = *(const v8bf*)sp;
        v8bf hi = *(const v8bf*)(sp + 16);
#pragma unroll
        for (int i = 0; i < 8; ++i) { a[i] = lo[i]; a[i + 8] = hi[i]; }
      }
      v16bf b0 = load_B32x16(kb + (size_t)t * C + cc * 32, C, lane);
      v16bf b1 = load_B32x16(kb + (size_t)(t + 16) * C + cc * 32, C, lane);
      P0 = wmma_bf16(a, b0, P0);
      P1 = wmma_bf16(a, b1, P1);
    }

    // ---- causal mask (keep t >= s); only the first chunk touches the diagonal ----
    if (t == t0) {
#pragma unroll
      for (int j = 0; j < 8; ++j) {
        const int s = s0 + mrow + j;
        if (t + ncol < s)      P0[j] = 0.0f;
        if (t + 16 + ncol < s) P1[j] = 0.0f;
      }
    }

    // ---- stage P (D layout) to LDS as 16x32 bf16 row-major, reload in A layout ----
#pragma unroll
    for (int j = 0; j < 8; ++j) {
      stage[wave][(mrow + j) * 32 + ncol]      = (bhalf)P0[j];
      stage[wave][(mrow + j) * 32 + 16 + ncol] = (bhalf)P1[j];
    }
    asm volatile("s_wait_dscnt 0" ::: "memory");
    v16bf ap;
    {
      const bhalf* sp = &stage[wave][arow * 32 + akof];
      v8bf lo = *(const v8bf*)sp;
      v8bf hi = *(const v8bf*)(sp + 16);
#pragma unroll
      for (int i = 0; i < 8; ++i) { ap[i] = lo[i]; ap[i + 8] = hi[i]; }
    }

    // ---- acc[s][c] += P(16x32) @ vT tiles ----
#pragma unroll
    for (int ct = 0; ct < 16; ++ct) {
      v16bf b = load_B32x16(vb + (size_t)(ct * 16) * NB + t, NB, lane);
      acc[ct] = wmma_bf16(ap, b, acc[ct]);
    }
  }

  // ---- tanh + bf16 store (row-major h) ----
#pragma unroll
  for (int ct = 0; ct < 16; ++ct)
#pragma unroll
    for (int j = 0; j < 8; ++j) {
      const float r = tanhf(acc[ct][j]);
      hout[((size_t)batch * NB + s0 + mrow + j) * C + ct * 16 + ncol] = (bhalf)r;
    }
}

// ---------- host ----------
extern "C" void kernel_launch(void* const* d_in, const int* in_sizes, int n_in,
                              void* d_out, int out_size, void* d_ws, size_t ws_size,
                              hipStream_t stream) {
  const float* x     = (const float*)d_in[0];  // (4,1,524288)
  const float* W_in  = (const float*)d_in[1];  // (256,128)
  const float* Wp    = (const float*)d_in[2];  // (4,256,256)
  const float* Wk    = (const float*)d_in[3];
  const float* Wv    = (const float*)d_in[4];
  const float* Wq    = (const float*)d_in[5];
  const float* W_out = (const float*)d_in[6];  // (128,256)
  float* out = (float*)d_out;                  // (4,4096,128)

  const int M = 16384;        // 4 batches * 4096 blocks
  char* ws = (char*)d_ws;
  auto alloc = [&](size_t bytes) -> char* {
    char* p = ws;
    ws += (bytes + 255) & ~(size_t)255;
    return p;
  };
  bhalf* xb    = (bhalf*)alloc((size_t)M * 128 * 2);
  bhalf* Winb  = (bhalf*)alloc((size_t)256 * 128 * 2);
  bhalf* Wpb   = (bhalf*)alloc((size_t)4 * 256 * 256 * 2);
  bhalf* Wkb   = (bhalf*)alloc((size_t)4 * 256 * 256 * 2);
  bhalf* Wvb   = (bhalf*)alloc((size_t)4 * 256 * 256 * 2);
  bhalf* Wqb   = (bhalf*)alloc((size_t)4 * 256 * 256 * 2);
  bhalf* Woutb = (bhalf*)alloc((size_t)128 * 256 * 2);
  bhalf* h     = (bhalf*)alloc((size_t)M * 256 * 2);
  bhalf* hp    = (bhalf*)alloc((size_t)M * 256 * 2);
  bhalf* kbuf  = (bhalf*)alloc((size_t)M * 256 * 2);
  bhalf* qbuf  = (bhalf*)alloc((size_t)M * 256 * 2);
  bhalf* vTb   = (bhalf*)alloc((size_t)M * 256 * 2);

  auto cvt = [&](const float* src, bhalf* dst, int n) {
    cvt_bf16_kernel<<<(n + 255) / 256, 256, 0, stream>>>(src, dst, n);
  };
  cvt(x, xb, M * 128);
  cvt(W_in, Winb, 256 * 128);
  cvt(Wp, Wpb, 4 * 256 * 256);
  cvt(Wk, Wkb, 4 * 256 * 256);
  cvt(Wv, Wvb, 4 * 256 * 256);
  cvt(Wq, Wqb, 4 * 256 * 256);
  cvt(W_out, Woutb, 128 * 256);

  // input projection: h = xb @ W_in^T   (K=128, N=256)
  gemm_bf16_kernel<16, 4, 0><<<M / 128, 256, 0, stream>>>(xb, Winb, h);

  for (int l = 0; l < 4; ++l) {
    const size_t wo = (size_t)l * 256 * 256;
    gemm_bf16_kernel<16, 8, 0><<<M / 128, 256, 0, stream>>>(h, Wpb + wo, hp);
    gemm_bf16_kernel<16, 8, 0><<<M / 128, 256, 0, stream>>>(hp, Wkb + wo, kbuf);
    gemm_bf16_kernel<16, 8, 0><<<M / 128, 256, 0, stream>>>(hp, Wqb + wo, qbuf);
    gemm_bf16_kernel<16, 8, 2><<<M / 128, 256, 0, stream>>>(hp, Wvb + wo, vTb);
    attn_kernel<<<dim3(32, 4), 256, 0, stream>>>(kbuf, qbuf, vTb, h);
  }

  // output projection: out = h @ W_out^T   (K=256, N=128, fp32 out)
  gemm_bf16_kernel<8, 8, 1><<<M / 128, 256, 0, stream>>>(h, Woutb, out);
}
